// GRU_13262859010769
// MI455X (gfx1250) — compile-verified, matrix operand order checked by reading, not run on hardware
//
#include <hip/hip_runtime.h>

// ---------------------------------------------------------------------------
// 2-layer GRU for MI455X (gfx1250).
//   S=512, B=64, D=H=1024. Math path: bf16 WMMA (v_wmma_f32_16x16x32_bf16),
//   fp32 accumulation + fp32 gate math. Input projections as big GEMMs;
//   the recurrence runs in a persistent kernel with a device-wide barrier
//   per timestep. W_hh slices live in LDS (192 KB/WGP) for the whole scan.
// ---------------------------------------------------------------------------

typedef __bf16 bf16_t;
typedef __attribute__((ext_vector_type(16))) __bf16 v16bf;
typedef __attribute__((ext_vector_type(8)))  __bf16 v8bf;
typedef __attribute__((ext_vector_type(8)))  float   v8f;

#define S_N    512
#define B_N    64
#define D_N    1024
#define H_N    1024
#define H3_N   3072
#define SCAN_BLOCKS 32           // 32 blocks * 8 waves = 256 waves = 256 tiles
#define SCAN_LDS_BYTES (3 * 32 * H_N * 2)   // 192 KB: 3 gates x 32 cols x K

static __device__ __forceinline__ v8f zero8() {
  v8f z = {0.f, 0.f, 0.f, 0.f, 0.f, 0.f, 0.f, 0.f};
  return z;
}

static __device__ __forceinline__ v8f wmma_bf16(v16bf a, v16bf b, v8f c) {
  // (neg_a, A, neg_b, B, c_mod, C, reuse_a, reuse_b)
  return __builtin_amdgcn_wmma_f32_16x16x32_bf16(false, a, false, b,
                                                 (short)0, c, false, false);
}

// NT fragment load per ISA 7.12.2 16-bit operand layout (A by rows, B by
// cols; both K-contiguous in memory). Two global_load_b128 / ds_load_b128.
//   lane<16 : row/col = rc + lane,    K in {k0..k0+7} U {k0+16..k0+23}
//   lane>=16: row/col = rc + lane-16, K in {k0+8..k0+15} U {k0+24..k0+31}
static __device__ __forceinline__ v16bf load_frag(const bf16_t* __restrict__ p0,
                                                  int ld, int rc, int k0, int lane) {
  const bf16_t* p = p0 + (size_t)(rc + (lane & 15)) * ld + (k0 + ((lane >> 4) << 3));
  v16bf f;
  ((v8bf*)&f)[0] = *(const v8bf*)(p);
  ((v8bf*)&f)[1] = *(const v8bf*)(p + 16);
  return f;
}

static __device__ __forceinline__ float sigmoidf_(float x) {
  return 1.f / (1.f + __expf(-x));
}
static __device__ __forceinline__ float tanhf_(float x) {
  float e = __expf(2.f * x);
  return 1.f - 2.f / (e + 1.f);
}

// Device-wide sense barrier: bar[0]=arrival count, bar[1]=generation.
// __threadfence() -> global_wb / global_inv at device scope around the
// atomic handshake so every wave sees the h state written by other WGPs.
static __device__ __forceinline__ void grid_barrier(unsigned* bar) {
  __threadfence();                 // flush this wave's h stores device-wide
  __syncthreads();
  if (threadIdx.x == 0) {
    unsigned* cnt = bar;
    unsigned* gen = bar + 1;
    const unsigned g = __hip_atomic_load(gen, __ATOMIC_RELAXED, __HIP_MEMORY_SCOPE_AGENT);
    const unsigned a = __hip_atomic_fetch_add(cnt, 1u, __ATOMIC_ACQ_REL, __HIP_MEMORY_SCOPE_AGENT);
    if (a == gridDim.x - 1) {
      __hip_atomic_store(cnt, 0u, __ATOMIC_RELAXED, __HIP_MEMORY_SCOPE_AGENT);
      __hip_atomic_store(gen, g + 1u, __ATOMIC_RELEASE, __HIP_MEMORY_SCOPE_AGENT);
    } else {
      while (__hip_atomic_load(gen, __ATOMIC_ACQUIRE, __HIP_MEMORY_SCOPE_AGENT) == g) {
        __builtin_amdgcn_s_sleep(2);
      }
    }
  }
  __syncthreads();
  __threadfence();                 // invalidate stale lines before next step
}

// ---------------------------------------------------------------------------
// fp32 -> bf16 conversion (4 elems / thread)
// ---------------------------------------------------------------------------
__global__ void cvt_f32_bf16_kernel(const float* __restrict__ in,
                                    bf16_t* __restrict__ out, long n) {
  long i = ((long)blockIdx.x * blockDim.x + threadIdx.x) * 4;
  if (i + 3 < n) {
    float4 v = *(const float4*)(in + i);
    bf16_t o4[4] = {(bf16_t)v.x, (bf16_t)v.y, (bf16_t)v.z, (bf16_t)v.w};
    *(uint2*)(out + i) = *(const uint2*)o4;
  } else {
    for (; i < n; ++i) out[i] = (bf16_t)in[i];
  }
}

// ---------------------------------------------------------------------------
// Projection GEMM: out(M,N) = A(M,K) * W(N,K)^T + bias(N), bf16 WMMA.
// Block = 8 waves -> 128x64 block tile; each wave does a 32x32 tile
// (2x2 WMMA tiles) with 2x A/B fragment reuse.
// ---------------------------------------------------------------------------
__global__ __launch_bounds__(256)
void proj_gemm_kernel(const bf16_t* __restrict__ A,
                      const bf16_t* __restrict__ W,
                      const float* __restrict__ bias,
                      float* __restrict__ out,
                      int M, int N, int K) {
  const int lane = threadIdx.x & 31;
  const int wid  = threadIdx.x >> 5;   // 0..7
  const int wm   = wid & 3;            // 4 waves along M
  const int wn   = wid >> 2;           // 2 waves along N
  const int m0 = blockIdx.y * 128 + wm * 32;
  const int n0 = blockIdx.x * 64 + wn * 32;

  v8f acc00 = zero8(), acc01 = zero8(), acc10 = zero8(), acc11 = zero8();
#pragma unroll 4
  for (int k0 = 0; k0 < K; k0 += 32) {
    v16bf a0 = load_frag(A, K, m0,      k0, lane);
    v16bf a1 = load_frag(A, K, m0 + 16, k0, lane);
    v16bf b0 = load_frag(W, K, n0,      k0, lane);
    v16bf b1 = load_frag(W, K, n0 + 16, k0, lane);
    acc00 = wmma_bf16(a0, b0, acc00);
    acc01 = wmma_bf16(a0, b1, acc01);
    acc10 = wmma_bf16(a1, b0, acc10);
    acc11 = wmma_bf16(a1, b1, acc11);
  }

  const int rrow = (lane >> 4) * 8;    // C layout: VGPR i -> M = i (+8 hi half)
  const int ccol = lane & 15;
  v8f accs[2][2] = {{acc00, acc01}, {acc10, acc11}};
#pragma unroll
  for (int tm = 0; tm < 2; ++tm) {
#pragma unroll
    for (int tn = 0; tn < 2; ++tn) {
      const int col = n0 + tn * 16 + ccol;
      const float bb = bias[col];
#pragma unroll
      for (int i = 0; i < 8; ++i) {
        const int row = m0 + tm * 16 + rrow + i;
        out[(size_t)row * N + col] = accs[tm][tn][i] + bb;
      }
    }
  }
}

// ---------------------------------------------------------------------------
// Persistent GRU recurrence. 32 blocks x 8 waves; each wave owns one 16x16
// h-tile (mb in [0,4), nb in [0,64)) and computes the r/z/n gate GEMMs for
// it each step. W_hh slice (3 gates x 32 cols x 1024 K, bf16) staged in LDS
// once. out[s] doubles as the h state: out_bf[s-1] feeds the next step's
// WMMA A-matrix, out_f32[s-1] the z*h term. Device-wide barrier per step.
// ---------------------------------------------------------------------------
__global__ __launch_bounds__(256, 1)
void gru_scan_kernel(const float* __restrict__ xlin,    // (S,B,3H) = x@W_ih^T + b_ih
                     const bf16_t* __restrict__ whh,    // (3H,H) bf16
                     const float* __restrict__ bhh,     // (3H)
                     const float* __restrict__ h0f,     // (B,H) fp32 initial h
                     const bf16_t* __restrict__ h0bf,   // (B,H) bf16 initial h
                     float* __restrict__ outf,          // (S,B,H)
                     bf16_t* __restrict__ outbf,        // (S,B,H)
                     unsigned* __restrict__ bar,
                     int steps) {
  extern __shared__ bf16_t lds_w[];    // [3][32][H_N]

  const int lane = threadIdx.x & 31;
  const int wid  = threadIdx.x >> 5;          // 0..7
  const int gw   = blockIdx.x * 8 + wid;      // 0..255 global wave
  const int mb   = gw & 3;                    // batch-row block
  const int nbl  = wid >> 2;                  // 0/1: local col block
  const int n_base = blockIdx.x * 32;         // block's first output column
  const int m0 = mb * 16;
  const int n0 = n_base + nbl * 16;
  const int c0 = nbl * 16;                    // LDS-local col base
  const int rrow = (lane >> 4) * 8;
  const int ccol = lane & 15;

  // Stage this block's W_hh slice into LDS (16B per thread per iteration).
  for (int t = threadIdx.x; t < (3 * 32 * H_N) / 8; t += 256) {
    const int idx = t * 8;
    const int k  = idx & (H_N - 1);
    const int rc = idx >> 10;                 // g*32 + c
    const int g  = rc >> 5;
    const int c  = rc & 31;
    *(v8bf*)(lds_w + idx) =
        *(const v8bf*)(whh + ((size_t)(g * H_N + n_base + c)) * H_N + k);
  }
  __syncthreads();

  const int col = n0 + ccol;
  const float br_ = bhh[col];
  const float bz_ = bhh[col + H_N];
  const float bn_ = bhh[col + 2 * H_N];

  for (int s = 0; s < steps; ++s) {
    const bf16_t* hsrc = (s == 0) ? h0bf : (outbf + (size_t)(s - 1) * B_N * H_N);

    v8f ar = zero8(), az = zero8(), an = zero8();
#pragma unroll 4
    for (int k0 = 0; k0 < H_N; k0 += 32) {
      v16bf a  = load_frag(hsrc, H_N, m0, k0, lane);           // global (L2)
      v16bf brf = load_frag(lds_w, H_N, 0 * 32 + c0, k0, lane); // LDS
      v16bf bzf = load_frag(lds_w, H_N, 1 * 32 + c0, k0, lane); // LDS
      v16bf bnf = load_frag(lds_w, H_N, 2 * 32 + c0, k0, lane); // LDS
      ar = wmma_bf16(a, brf, ar);
      az = wmma_bf16(a, bzf, az);
      an = wmma_bf16(a, bnf, an);
    }

    const float* hop = (s == 0) ? h0f : (outf + (size_t)(s - 1) * B_N * H_N);
#pragma unroll
    for (int i = 0; i < 8; ++i) {
      const int row = m0 + rrow + i;
      const float* xb = xlin + ((size_t)s * B_N + row) * H3_N;
      const float r  = sigmoidf_(ar[i] + xb[col] + br_);
      const float z  = sigmoidf_(az[i] + xb[col + H_N] + bz_);
      const float nn = tanhf_(xb[col + 2 * H_N] + r * (an[i] + bn_));
      const float hold = hop[(size_t)row * H_N + col];
      const float hnew = (1.f - z) * nn + z * hold;
      const size_t o = ((size_t)s * B_N + row) * H_N + col;
      outf[o]  = hnew;
      outbf[o] = (bf16_t)hnew;
    }
    grid_barrier(bar);   // publish h(s) before anyone starts step s+1
  }
}

// ---------------------------------------------------------------------------
// Final hidden states: hidden[0]=out0[S-1], hidden[1]=out1[S-1].
// ---------------------------------------------------------------------------
__global__ void hidden_copy_kernel(const float* __restrict__ out0_last,
                                   const float* __restrict__ out1_last,
                                   float* __restrict__ dst, int n) {
  int i = blockIdx.x * blockDim.x + threadIdx.x;
  if (i < n) {
    dst[i]     = out0_last[i];
    dst[n + i] = out1_last[i];
  }
}

// ---------------------------------------------------------------------------
extern "C" void kernel_launch(void* const* d_in, const int* in_sizes, int n_in,
                              void* d_out, int out_size, void* d_ws, size_t ws_size,
                              hipStream_t stream) {
  const float* x     = (const float*)d_in[0];
  const float* hx    = (const float*)d_in[1];
  const float* wih0  = (const float*)d_in[2];
  const float* whh0  = (const float*)d_in[3];
  const float* bih0  = (const float*)d_in[4];
  const float* bhh0  = (const float*)d_in[5];
  const float* wih1  = (const float*)d_in[6];
  const float* whh1  = (const float*)d_in[7];
  const float* bih1  = (const float*)d_in[8];
  const float* bhh1  = (const float*)d_in[9];
  float* out = (float*)d_out;

  const size_t M    = (size_t)S_N * B_N;          // 32768 GEMM rows
  const size_t nX   = M * D_N;                    // x elements
  const size_t nW   = (size_t)H3_N * H_N;         // per weight matrix
  const size_t nH   = (size_t)B_N * H_N;          // per-layer h state
  const size_t nOut = M * H_N;                    // per-layer sequence output

  // Workspace carve-up (256B aligned).
  size_t off = 0;
  auto carve = [&](size_t bytes) -> char* {
    char* p = (char*)d_ws + off;
    off = (off + bytes + 255) & ~(size_t)255;
    return p;
  };
  float*  xlin    = (float*)carve(M * H3_N * sizeof(float));   // 384 MB (reused L0/L1)
  float*  out0f   = (float*)carve(nOut * sizeof(float));       // 128 MB
  bf16_t* seq_bf  = (bf16_t*)carve(nOut * sizeof(bf16_t));     //  64 MB (L0 out, then L1 state)
  bf16_t* x_bf    = (bf16_t*)carve(nX * sizeof(bf16_t));       //  64 MB
  bf16_t* wih0_bf = (bf16_t*)carve(nW * sizeof(bf16_t));
  bf16_t* whh0_bf = (bf16_t*)carve(nW * sizeof(bf16_t));
  bf16_t* wih1_bf = (bf16_t*)carve(nW * sizeof(bf16_t));
  bf16_t* whh1_bf = (bf16_t*)carve(nW * sizeof(bf16_t));
  bf16_t* h0bf0   = (bf16_t*)carve(nH * sizeof(bf16_t));
  bf16_t* h0bf1   = (bf16_t*)carve(nH * sizeof(bf16_t));
  unsigned* bar   = (unsigned*)carve(256);
  if (ws_size < off) return;   // insufficient scratch; nothing safe to do

  hipMemsetAsync(bar, 0, 256, stream);

  // fp32 -> bf16 staging.
  const int CT = 256;
  auto cvt = [&](const float* src, bf16_t* dst, long n) {
    long thr = (n + 3) / 4;
    cvt_f32_bf16_kernel<<<(unsigned)((thr + CT - 1) / CT), CT, 0, stream>>>(src, dst, n);
  };
  cvt(x,    x_bf,    (long)nX);
  cvt(wih0, wih0_bf, (long)nW);
  cvt(whh0, whh0_bf, (long)nW);
  cvt(wih1, wih1_bf, (long)nW);
  cvt(whh1, whh1_bf, (long)nW);
  cvt(hx,            h0bf0, (long)nH);
  cvt(hx + nH,       h0bf1, (long)nH);

  dim3 pgrid(H3_N / 64, (unsigned)(M / 128));   // 48 x 256 blocks

  // ---- Layer 0 ----
  proj_gemm_kernel<<<pgrid, 256, 0, stream>>>(x_bf, wih0_bf, bih0, xlin,
                                              (int)M, H3_N, D_N);
  gru_scan_kernel<<<SCAN_BLOCKS, 256, SCAN_LDS_BYTES, stream>>>(
      xlin, whh0_bf, bhh0, hx, h0bf0, out0f, seq_bf, bar, S_N);

  // ---- Layer 1 (A-matrix = layer-0 bf16 output; xlin buffer reused) ----
  proj_gemm_kernel<<<pgrid, 256, 0, stream>>>(seq_bf, wih1_bf, bih1, xlin,
                                              (int)M, H3_N, H_N);
  gru_scan_kernel<<<SCAN_BLOCKS, 256, SCAN_LDS_BYTES, stream>>>(
      xlin, whh1_bf, bhh1, hx + nH, h0bf1, out, seq_bf, bar, S_N);

  // ---- hidden = stack(out0[S-1], out1[S-1]) appended after out1 ----
  hidden_copy_kernel<<<(unsigned)((nH + 255) / 256), 256, 0, stream>>>(
      out0f + (size_t)(S_N - 1) * nH, out + (size_t)(S_N - 1) * nH,
      out + nOut, (int)nH);
}
